// Mamba_hugginface_80255758893799
// MI455X (gfx1250) — compile-verified, matrix operand order checked by reading, not run on hardware
//
#include <hip/hip_runtime.h>
#include <hip/hip_bf16.h>
#include <cstdint>
#include <cstddef>

// ---------------- Model constants (from reference) ----------------
#define V_    32000
#define D_    768
#define L_    4
#define N_    16
#define KCONV 4
#define I_    1536
#define DT_   48
#define XP_   80          // DT + 2N
#define B_    4
#define S_    2048
#define BS_   (B_*S_)
#define EPS_  1e-5f

// ---------------- WMMA tile config ----------------
#define TILE_M 128
#define TILE_N 64
#define TILE_K 64         // two 32-K WMMA chunks per staged tile
#define LDSTR  72         // halves per LDS row (64 data + 8 pad) = 144B (9x16B)

typedef __attribute__((ext_vector_type(16))) __bf16 v16bf;
typedef __attribute__((ext_vector_type(8)))  float  v8f;
typedef unsigned short ushort_t;

union FragBF {
    uint4 q[2];   // 2 x 16B chunks from LDS
    v16bf v;      // WMMA operand
};

__device__ __forceinline__ ushort_t f2bf(float f) {
    unsigned int u = __float_as_uint(f);
    u += 0x7FFFu + ((u >> 16) & 1u);          // round-to-nearest-even
    return (ushort_t)(u >> 16);
}
__device__ __forceinline__ unsigned int pack2bf(float a, float b) {
    unsigned int ua = __float_as_uint(a);
    unsigned int ub = __float_as_uint(b);
    ua += 0x7FFFu + ((ua >> 16) & 1u);
    ub += 0x7FFFu + ((ub >> 16) & 1u);
    return (ua >> 16) | (ub & 0xFFFF0000u);
}
__device__ __forceinline__ float softplusf(float x) {
    return (x > 20.f) ? x : log1pf(__expf(x));
}
__device__ __forceinline__ float siluf(float x) {
    return x / (1.f + __expf(-x));
}

// --- CDNA5 async global->LDS copy (ASYNCcnt tracked), per-lane 16B ---
__device__ __forceinline__ void async_copy_b128(unsigned lds_off, unsigned long long gaddr) {
    asm volatile("global_load_async_to_lds_b128 %0, %1, off"
                 :: "v"(lds_off), "v"(gaddr) : "memory");
}
__device__ __forceinline__ void wait_asynccnt0() {
    asm volatile("s_wait_asynccnt 0x0" ::: "memory");
}
// Each wave issues exactly 6 async b128 ops per tile (4 A + 2 W); waiting <=6
// drains the previous tile while the just-issued next tile stays in flight.
__device__ __forceinline__ void wait_asynccnt6() {
    asm volatile("s_wait_asynccnt 0x6" ::: "memory");
}

__device__ __forceinline__ FragBF load_frag(const ushort_t* buf, int row, int kk, int g) {
    // lane l+16g holds matrix row `row`; within K-chunk kk: halves [8g..] and [16+8g..]
    const ushort_t* p = buf + row * LDSTR + kk * 32;
    FragBF f;
    f.q[0] = *(const uint4*)(p + 8 * g);
    f.q[1] = *(const uint4*)(p + 16 + 8 * g);
    return f;
}

// ---------------------------------------------------------------------------
// Generic NT GEMM, bf16 WMMA, f32 accumulate:
//   C[M,Nc] = A[M,Kd] (bf16, lda) x W[Nc,Kd]^T (bf16 or f32, ldb)
// Fast path: full tiles + Kd%64==0 + bf16 W -> double-buffered
// global_load_async_to_lds_b128 pipeline with partial ASYNCcnt waits.
// Fallback: branch-minimal clamp+select VALU staging.
// mode 0: store; 1: C += acc; 2: C = softplus(acc + bias[n]).
// Cbf (optional) mirrors result as bf16.
// ---------------------------------------------------------------------------
__global__ __launch_bounds__(256) void wmma_gemm_nt(
    const ushort_t* __restrict__ A, const void* __restrict__ Wp, int w_bf16,
    float* __restrict__ Cm, ushort_t* __restrict__ Cbf, const float* __restrict__ bias,
    int M, int Nc, int Kd, int lda, int ldb, int ldc, int mode)
{
    __shared__ __align__(16) ushort_t As[2][TILE_M * LDSTR];
    __shared__ __align__(16) ushort_t Ws[2][TILE_N * LDSTR];

    const int tid  = threadIdx.x;
    const int lane = tid & 31;
    const int wave = tid >> 5;         // 0..7
    const int wm   = wave >> 1;        // 0..3  (M dir, 32 rows each)
    const int wn   = wave & 1;         // 0..1  (N dir, 32 cols each)
    const int lcol = lane & 15;
    const int grp  = lane >> 4;

    const int blockM = blockIdx.y * TILE_M;
    const int blockN = blockIdx.x * TILE_N;

    v8f acc[2][2] = {};

    const bool fastpath = ((blockM + TILE_M) <= M) && ((blockN + TILE_N) <= Nc)
                          && w_bf16 && ((Kd & (TILE_K - 1)) == 0);

    if (fastpath) {
        const ushort_t* Wb = (const ushort_t*)Wp;
        // per-thread async roles: A = 4 chunks, W = 2 chunks (16B each)
        const int T = Kd / TILE_K;
        int cur = 0;

        // ---- issue tile 0 into buffer 0 ----
        {
            const unsigned aB = (unsigned)(uintptr_t)(&As[0][0]);
            const unsigned wB = (unsigned)(uintptr_t)(&Ws[0][0]);
            #pragma unroll
            for (int q = 0; q < 4; ++q) {
                const int c = tid + q * 256;           // 0..1023
                const int row = c >> 3, ch = c & 7;    // 128 rows x 8 chunks
                async_copy_b128(aB + (unsigned)(row * (LDSTR * 2) + ch * 16),
                    (unsigned long long)(const void*)(A + (size_t)(blockM + row) * lda + ch * 8));
            }
            #pragma unroll
            for (int q = 0; q < 2; ++q) {
                const int c = tid + q * 256;           // 0..511
                const int row = c >> 3, ch = c & 7;    // 64 rows x 8 chunks
                async_copy_b128(wB + (unsigned)(row * (LDSTR * 2) + ch * 16),
                    (unsigned long long)(const void*)(Wb + (size_t)(blockN + row) * ldb + ch * 8));
            }
        }

        for (int t = 0; t < T; ++t) {
            if (t + 1 < T) {
                const int k1 = (t + 1) * TILE_K;
                const unsigned aB = (unsigned)(uintptr_t)(&As[cur ^ 1][0]);
                const unsigned wB = (unsigned)(uintptr_t)(&Ws[cur ^ 1][0]);
                #pragma unroll
                for (int q = 0; q < 4; ++q) {
                    const int c = tid + q * 256;
                    const int row = c >> 3, ch = c & 7;
                    async_copy_b128(aB + (unsigned)(row * (LDSTR * 2) + ch * 16),
                        (unsigned long long)(const void*)(A + (size_t)(blockM + row) * lda + k1 + ch * 8));
                }
                #pragma unroll
                for (int q = 0; q < 2; ++q) {
                    const int c = tid + q * 256;
                    const int row = c >> 3, ch = c & 7;
                    async_copy_b128(wB + (unsigned)(row * (LDSTR * 2) + ch * 16),
                        (unsigned long long)(const void*)(Wb + (size_t)(blockN + row) * ldb + k1 + ch * 8));
                }
                wait_asynccnt6();    // previous tile complete; next 6 stay in flight
            } else {
                wait_asynccnt0();
            }
            __syncthreads();

            const ushort_t* AsB = &As[cur][0];
            const ushort_t* WsB = &Ws[cur][0];
            #pragma unroll
            for (int kk = 0; kk < 2; ++kk) {
                FragBF a0 = load_frag(AsB, wm * 32 + lcol,      kk, grp);
                FragBF a1 = load_frag(AsB, wm * 32 + 16 + lcol, kk, grp);
                FragBF b0 = load_frag(WsB, wn * 32 + lcol,      kk, grp);
                FragBF b1 = load_frag(WsB, wn * 32 + 16 + lcol, kk, grp);
                acc[0][0] = __builtin_amdgcn_wmma_f32_16x16x32_bf16(false, a0.v, false, b0.v, (short)0, acc[0][0], false, false);
                acc[0][1] = __builtin_amdgcn_wmma_f32_16x16x32_bf16(false, a0.v, false, b1.v, (short)0, acc[0][1], false, false);
                acc[1][0] = __builtin_amdgcn_wmma_f32_16x16x32_bf16(false, a1.v, false, b0.v, (short)0, acc[1][0], false, false);
                acc[1][1] = __builtin_amdgcn_wmma_f32_16x16x32_bf16(false, a1.v, false, b1.v, (short)0, acc[1][1], false, false);
            }
            __syncthreads();
            cur ^= 1;
        }
    } else {
        // -------------------- fallback: clamp+select VALU staging --------------------
        const int arow = tid >> 1;             // A: 128 rows, kb = (tid&1)*32 halves
        const int akb  = (tid & 1) * 32;
        const int wrow = tid >> 2;             // W: 64 rows, kb = (tid&3)*16 halves
        const int wkb  = (tid & 3) * 16;

        for (int k0 = 0; k0 < Kd; k0 += TILE_K) {
            // ---- A tile ----
            {
                const int rowG = blockM + arow;
                const int rc   = rowG < M ? rowG : (M - 1);
                const size_t rb = (size_t)rc * lda;
                #pragma unroll
                for (int half = 0; half < 4; ++half) {   // 4 x uint4 = 32 halves
                    uint4 s;
                    unsigned pk[4];
                    #pragma unroll
                    for (int q = 0; q < 4; ++q) {
                        const int kg0 = k0 + akb + half * 8 + 2 * q;
                        const int kg1 = kg0 + 1;
                        const int kc0 = kg0 < Kd ? kg0 : (Kd - 1);
                        const int kc1 = kg1 < Kd ? kg1 : (Kd - 1);
                        unsigned h0 = A[rb + kc0];
                        unsigned h1 = A[rb + kc1];
                        h0 = (rowG < M && kg0 < Kd) ? h0 : 0u;
                        h1 = (rowG < M && kg1 < Kd) ? h1 : 0u;
                        pk[q] = h0 | (h1 << 16);
                    }
                    s.x = pk[0]; s.y = pk[1]; s.z = pk[2]; s.w = pk[3];
                    *(uint4*)&As[0][arow * LDSTR + akb + half * 8] = s;
                }
            }
            // ---- W tile ----
            if (w_bf16) {
                const ushort_t* Wb = (const ushort_t*)Wp;
                const int rowG = blockN + wrow;
                const int rc   = rowG < Nc ? rowG : (Nc - 1);
                const size_t rb = (size_t)rc * ldb;
                #pragma unroll
                for (int half = 0; half < 2; ++half) {
                    uint4 s;
                    unsigned pk[4];
                    #pragma unroll
                    for (int q = 0; q < 4; ++q) {
                        const int kg0 = k0 + wkb + half * 8 + 2 * q;
                        const int kg1 = kg0 + 1;
                        const int kc0 = kg0 < Kd ? kg0 : (Kd - 1);
                        const int kc1 = kg1 < Kd ? kg1 : (Kd - 1);
                        unsigned h0 = Wb[rb + kc0];
                        unsigned h1 = Wb[rb + kc1];
                        h0 = (rowG < Nc && kg0 < Kd) ? h0 : 0u;
                        h1 = (rowG < Nc && kg1 < Kd) ? h1 : 0u;
                        pk[q] = h0 | (h1 << 16);
                    }
                    s.x = pk[0]; s.y = pk[1]; s.z = pk[2]; s.w = pk[3];
                    *(uint4*)&Ws[0][wrow * LDSTR + wkb + half * 8] = s;
                }
            } else {
                const float* Wf = (const float*)Wp;
                const int rowG = blockN + wrow;
                const int rc   = rowG < Nc ? rowG : (Nc - 1);
                const size_t rb = (size_t)rc * ldb;
                #pragma unroll
                for (int half = 0; half < 2; ++half) {
                    uint4 s;
                    unsigned pk[4];
                    #pragma unroll
                    for (int q = 0; q < 4; ++q) {
                        const int kg0 = k0 + wkb + half * 8 + 2 * q;
                        const int kg1 = kg0 + 1;
                        const int kc0 = kg0 < Kd ? kg0 : (Kd - 1);
                        const int kc1 = kg1 < Kd ? kg1 : (Kd - 1);
                        float v0 = Wf[rb + kc0];
                        float v1 = Wf[rb + kc1];
                        v0 = (rowG < Nc && kg0 < Kd) ? v0 : 0.f;
                        v1 = (rowG < Nc && kg1 < Kd) ? v1 : 0.f;
                        pk[q] = pack2bf(v0, v1);
                    }
                    s.x = pk[0]; s.y = pk[1]; s.z = pk[2]; s.w = pk[3];
                    *(uint4*)&Ws[0][wrow * LDSTR + wkb + half * 8] = s;
                }
            }
            __syncthreads();

            #pragma unroll
            for (int kk = 0; kk < 2; ++kk) {
                FragBF a0 = load_frag(&As[0][0], wm * 32 + lcol,      kk, grp);
                FragBF a1 = load_frag(&As[0][0], wm * 32 + 16 + lcol, kk, grp);
                FragBF b0 = load_frag(&Ws[0][0], wn * 32 + lcol,      kk, grp);
                FragBF b1 = load_frag(&Ws[0][0], wn * 32 + 16 + lcol, kk, grp);
                acc[0][0] = __builtin_amdgcn_wmma_f32_16x16x32_bf16(false, a0.v, false, b0.v, (short)0, acc[0][0], false, false);
                acc[0][1] = __builtin_amdgcn_wmma_f32_16x16x32_bf16(false, a0.v, false, b1.v, (short)0, acc[0][1], false, false);
                acc[1][0] = __builtin_amdgcn_wmma_f32_16x16x32_bf16(false, a1.v, false, b0.v, (short)0, acc[1][0], false, false);
                acc[1][1] = __builtin_amdgcn_wmma_f32_16x16x32_bf16(false, a1.v, false, b1.v, (short)0, acc[1][1], false, false);
            }
            __syncthreads();
        }
    }

    // ---- epilogue (C/D layout: lane l+16g, VGPR r -> (M=r+8g, N=l)) ----
    #pragma unroll
    for (int i2 = 0; i2 < 2; ++i2) {
        #pragma unroll
        for (int j2 = 0; j2 < 2; ++j2) {
            const int col = blockN + wn * 32 + j2 * 16 + lcol;
            if (col >= Nc) continue;
            const int rbase = blockM + wm * 32 + i2 * 16 + grp * 8;
            #pragma unroll
            for (int r = 0; r < 8; ++r) {
                const int row = rbase + r;
                if (row >= M) continue;
                const size_t o = (size_t)row * ldc + col;
                const float v = acc[i2][j2][r];
                float outv;
                if (mode == 1)      { outv = Cm[o] + v; Cm[o] = outv; }
                else if (mode == 2) { outv = softplusf(v + bias[col]); Cm[o] = outv; }
                else                { outv = v; Cm[o] = v; }
                if (Cbf) Cbf[o] = f2bf(outv);
            }
        }
    }
}

// ---------------------------------------------------------------------------
// f32 -> bf16 bulk conversion (weights, once per call)
// ---------------------------------------------------------------------------
__global__ __launch_bounds__(256) void f32_to_bf16_kernel(
    const float* __restrict__ src, ushort_t* __restrict__ dst, long n)
{
    const long i = ((long)blockIdx.x * 256 + threadIdx.x) * 4;
    if (i + 3 < n) {
        float4 f = *(const float4*)(src + i);
        uint2 p;
        p.x = pack2bf(f.x, f.y);
        p.y = pack2bf(f.z, f.w);
        *(uint2*)(dst + i) = p;
    } else {
        for (long q = i; q < n; ++q) dst[q] = f2bf(src[q]);
    }
}

// ---------------------------------------------------------------------------
// Embedding gather: h[row, :] = embed[ids[row], :]   (float4 granularity)
// ---------------------------------------------------------------------------
__global__ __launch_bounds__(256) void embed_kernel(
    const int* __restrict__ ids, const float* __restrict__ embed, float* __restrict__ h)
{
    const int idx = blockIdx.x * 256 + threadIdx.x;
    const int total = BS_ * (D_ / 4);
    if (idx >= total) return;
    const int dq  = idx % (D_ / 4);
    const int row = idx / (D_ / 4);
    const int tok = ids[row];
    ((float4*)h)[(size_t)row * (D_ / 4) + dq] =
        ((const float4*)embed)[(size_t)tok * (D_ / 4) + dq];
}

// ---------------------------------------------------------------------------
// RMSNorm producing bf16 rows (GEMM A operand); strided rows for last-token.
// ---------------------------------------------------------------------------
__global__ __launch_bounds__(256) void rmsnorm_bf16_kernel(
    const float* __restrict__ x, const float* __restrict__ w,
    ushort_t* __restrict__ out, long in_stride)
{
    __shared__ float red[8];
    const int row = blockIdx.x;
    const float* xr = x + (size_t)row * in_stride;
    float ss = 0.f;
    for (int d = threadIdx.x; d < D_; d += 256) { float v = xr[d]; ss += v * v; }
    #pragma unroll
    for (int off = 16; off > 0; off >>= 1) ss += __shfl_down(ss, off, 32);
    if ((threadIdx.x & 31) == 0) red[threadIdx.x >> 5] = ss;
    __syncthreads();
    if (threadIdx.x == 0) {
        float t = 0.f;
        #pragma unroll
        for (int q = 0; q < 8; ++q) t += red[q];
        red[0] = rsqrtf(t / (float)D_ + EPS_);
    }
    __syncthreads();
    const float r = red[0];
    for (int d = threadIdx.x; d < D_; d += 256)
        out[(size_t)row * D_ + d] = f2bf(xr[d] * r * w[d]);
}

// ---------------------------------------------------------------------------
// Depthwise causal conv (K=4) + bias + SiLU. Writes f32 (scan) + bf16 (GEMM A).
// ---------------------------------------------------------------------------
__global__ __launch_bounds__(256) void conv_silu_kernel(
    const float* __restrict__ proj, const float* __restrict__ cw,
    const float* __restrict__ cb, float* __restrict__ utf, ushort_t* __restrict__ utb)
{
    const int idx = blockIdx.x * 256 + threadIdx.x;
    if (idx >= B_ * S_ * I_) return;
    const int i  = idx % I_;
    const int s  = (idx / I_) % S_;
    const int bb = idx / (I_ * S_);
    float acc = cb[i];
    const float4 wv = *(const float4*)(cw + i * KCONV);
    const float wk[4] = { wv.x, wv.y, wv.z, wv.w };
    #pragma unroll
    for (int k = 0; k < KCONV; ++k) {
        const int sp = s - (KCONV - 1) + k;
        if (sp >= 0)
            acc += proj[((size_t)(bb * S_ + sp)) * (2 * I_) + i] * wk[k];
    }
    const float v = siluf(acc);
    utf[idx] = v;
    utb[idx] = f2bf(v);
}

// ---------------------------------------------------------------------------
// Selective scan: one thread per (batch, channel), N=16 state in VGPRs.
// B/C staged in LDS (64 steps/chunk). Fused: y = (scan_y + D*u)*silu(gate) -> bf16
// ---------------------------------------------------------------------------
__global__ __launch_bounds__(256) void mamba_scan_kernel(
    const float* __restrict__ xdbl,   // [BS, 80]: [.,48:64)=B, [.,64:80)=C
    const float* __restrict__ dt,     // [BS, I]
    const float* __restrict__ ut,     // [BS, I]
    const float* __restrict__ proj,   // [BS, 2I] (gate at +I)
    const float* __restrict__ a_log,  // [I, N]
    const float* __restrict__ d_par,  // [I]
    ushort_t* __restrict__ y)         // [BS, I] bf16
{
    __shared__ float bc[64][32];
    const int tid = threadIdx.x;
    const int b   = blockIdx.x / (I_ / 256);
    const int i   = (blockIdx.x % (I_ / 256)) * 256 + tid;

    float Av[N_];
    #pragma unroll
    for (int n = 0; n < N_; ++n) Av[n] = -__expf(a_log[(size_t)i * N_ + n]);
    const float dp = d_par[i];

    float hs[N_];
    #pragma unroll
    for (int n = 0; n < N_; ++n) hs[n] = 0.f;

    for (int c = 0; c < S_ / 64; ++c) {
        const int s0 = c * 64;
        __syncthreads();  // protect previous chunk's reads
        for (int e = tid; e < 64 * 32; e += 256) {
            const int t = e >> 5, j = e & 31;
            bc[t][j] = xdbl[((size_t)(b * S_ + s0 + t)) * XP_ + DT_ + j];
        }
        __syncthreads();
        for (int t = 0; t < 64; ++t) {
            const size_t rowb = (size_t)(b * S_ + s0 + t);
            const float dtv = dt[rowb * I_ + i];
            const float uv  = ut[rowb * I_ + i];
            const float du  = dtv * uv;
            float yv = 0.f;
            #pragma unroll
            for (int n = 0; n < N_; ++n) {
                const float dA = __expf(dtv * Av[n]);
                hs[n] = dA * hs[n] + du * bc[t][n];
                yv += hs[n] * bc[t][16 + n];
            }
            const float gv = proj[rowb * (size_t)(2 * I_) + I_ + i];
            y[rowb * I_ + i] = f2bf((yv + dp * uv) * siluf(gv));
        }
    }
}

// ---------------------------------------------------------------------------
// Host launcher. Workspace ~311 MB: f32 {h, proj, ut, dt, xdbl},
// bf16 {y/xn (aliased), ut, xdbl, hn, converted weights}.
// ---------------------------------------------------------------------------
extern "C" void kernel_launch(void* const* d_in, const int* in_sizes, int n_in,
                              void* d_out, int out_size, void* d_ws, size_t ws_size,
                              hipStream_t stream) {
    (void)in_sizes; (void)n_in; (void)out_size; (void)ws_size;
    const int*   ids       = (const int*)  d_in[0];
    const float* embed     = (const float*)d_in[1];
    const float* norm_w    = (const float*)d_in[2];   // (L, D)
    const float* in_proj_w = (const float*)d_in[3];   // (L, 2I, D)
    const float* conv_w    = (const float*)d_in[4];   // (L, I, K)
    const float* conv_b    = (const float*)d_in[5];   // (L, I)
    const float* x_proj_w  = (const float*)d_in[6];   // (L, 80, I)
    const float* dt_proj_w = (const float*)d_in[7];   // (L, I, DT)
    const float* dt_proj_b = (const float*)d_in[8];   // (L, I)
    const float* A_log     = (const float*)d_in[9];   // (L, I, N)
    const float* D_param   = (const float*)d_in[10];  // (L, I)
    const float* out_proj_w= (const float*)d_in[11];  // (L, D, I)
    const float* norm_f_w  = (const float*)d_in[12];  // (D,)

    // ---- workspace carve-up (256B aligned regions) ----
    char* base = (char*)d_ws;
    size_t off = 0;
    auto carve = [&](size_t bytes) {
        char* p = base + off;
        off += (bytes + 255) & ~(size_t)255;
        return p;
    };
    float*    h     = (float*)   carve((size_t)BS_ * D_ * 4);
    float*    proj  = (float*)   carve((size_t)BS_ * 2 * I_ * 4);
    float*    utf   = (float*)   carve((size_t)BS_ * I_ * 4);
    float*    dtb   = (float*)   carve((size_t)BS_ * I_ * 4);
    float*    xdbl  = (float*)   carve((size_t)BS_ * XP_ * 4);
    ushort_t* ynb   = (ushort_t*)carve((size_t)BS_ * I_ * 2);   // y (scan out) / xn alias
    ushort_t* utb   = (ushort_t*)carve((size_t)BS_ * I_ * 2);
    ushort_t* xdblb = (ushort_t*)carve((size_t)BS_ * XP_ * 2);
    ushort_t* hnb   = (ushort_t*)carve((size_t)B_ * D_ * 2);
    ushort_t* ipwb  = (ushort_t*)carve((size_t)L_ * 2 * I_ * D_ * 2);
    ushort_t* xpwb  = (ushort_t*)carve((size_t)L_ * XP_ * I_ * 2);
    ushort_t* dpwb  = (ushort_t*)carve((size_t)L_ * I_ * DT_ * 2);
    ushort_t* opwb  = (ushort_t*)carve((size_t)L_ * D_ * I_ * 2);
    ushort_t* xnb   = ynb;  // xn consumed by in_proj GEMM before scan writes y

    // ---- one-time (per call) weight conversion f32 -> bf16 ----
    {
        long n1 = (long)L_ * 2 * I_ * D_;
        long n2 = (long)L_ * XP_ * I_;
        long n3 = (long)L_ * I_ * DT_;
        long n4 = (long)L_ * D_ * I_;
        f32_to_bf16_kernel<<<(int)((n1 + 1023) / 1024), 256, 0, stream>>>(in_proj_w,  ipwb, n1);
        f32_to_bf16_kernel<<<(int)((n2 + 1023) / 1024), 256, 0, stream>>>(x_proj_w,   xpwb, n2);
        f32_to_bf16_kernel<<<(int)((n3 + 1023) / 1024), 256, 0, stream>>>(dt_proj_w,  dpwb, n3);
        f32_to_bf16_kernel<<<(int)((n4 + 1023) / 1024), 256, 0, stream>>>(out_proj_w, opwb, n4);
    }

    // ---- embedding gather ----
    embed_kernel<<<(BS_ * (D_ / 4) + 255) / 256, 256, 0, stream>>>(ids, embed, h);

    for (int l = 0; l < L_; ++l) {
        const float*    nw  = norm_w    + (size_t)l * D_;
        const ushort_t* ipw = ipwb      + (size_t)l * 2 * I_ * D_;
        const float*    cw  = conv_w    + (size_t)l * I_ * KCONV;
        const float*    cb  = conv_b    + (size_t)l * I_;
        const ushort_t* xpw = xpwb      + (size_t)l * XP_ * I_;
        const ushort_t* dpw = dpwb      + (size_t)l * I_ * DT_;
        const float*    dpb = dt_proj_b + (size_t)l * I_;
        const float*    al  = A_log     + (size_t)l * I_ * N_;
        const float*    dpr = D_param   + (size_t)l * I_;
        const ushort_t* opw = opwb      + (size_t)l * D_ * I_;

        // xn = rmsnorm(h, nw)  (bf16)
        rmsnorm_bf16_kernel<<<BS_, 256, 0, stream>>>(h, nw, xnb, (long)D_);

        // proj = xn @ ipw^T   [BS, 2I] f32
        {
            dim3 g((2 * I_ + TILE_N - 1) / TILE_N, (BS_ + TILE_M - 1) / TILE_M);
            wmma_gemm_nt<<<g, 256, 0, stream>>>(xnb, ipw, 1, proj, nullptr, nullptr,
                BS_, 2 * I_, D_, D_, D_, 2 * I_, 0);
        }

        // ut = silu(causal_conv(u) + cb)  (f32 + bf16)
        conv_silu_kernel<<<(B_ * S_ * I_ + 255) / 256, 256, 0, stream>>>(proj, cw, cb, utf, utb);

        // xdbl = ut @ xpw^T  [BS, 80] f32, mirrored bf16 for dt_proj A
        {
            dim3 g((XP_ + TILE_N - 1) / TILE_N, (BS_ + TILE_M - 1) / TILE_M);
            wmma_gemm_nt<<<g, 256, 0, stream>>>(utb, xpw, 1, xdbl, xdblb, nullptr,
                BS_, XP_, I_, I_, I_, XP_, 0);
        }

        // dt = softplus(xdbl[:, :48] @ dpw^T + dpb)  [BS, I] f32
        {
            dim3 g((I_ + TILE_N - 1) / TILE_N, (BS_ + TILE_M - 1) / TILE_M);
            wmma_gemm_nt<<<g, 256, 0, stream>>>(xdblb, dpw, 1, dtb, nullptr, dpb,
                BS_, I_, DT_, XP_, DT_, I_, 2);
        }

        // selective scan + fused (y + D*u)*silu(gate) -> bf16
        mamba_scan_kernel<<<B_ * (I_ / 256), 256, 0, stream>>>(
            xdbl, dtb, utf, proj, al, dpr, ynb);

        // h += y @ opw^T   (residual accumulate, f32)
        {
            dim3 g((D_ + TILE_N - 1) / TILE_N, (BS_ + TILE_M - 1) / TILE_M);
            wmma_gemm_nt<<<g, 256, 0, stream>>>(ynb, opw, 1, h, nullptr, nullptr,
                BS_, D_, I_, I_, I_, D_, 1);
        }
    }

    // final norm on last tokens (rows strided by S*D), then logits = hn @ embed^T
    rmsnorm_bf16_kernel<<<B_, 256, 0, stream>>>(h + (size_t)(S_ - 1) * D_, norm_f_w, hnb,
                                                (long)S_ * D_);
    {
        dim3 g((V_ + TILE_N - 1) / TILE_N, (B_ + TILE_M - 1) / TILE_M);
        wmma_gemm_nt<<<g, 256, 0, stream>>>(hnb, embed, 0, (float*)d_out, nullptr, nullptr,
            B_, V_, D_, D_, D_, V_, 0);
    }
}
